// NeuralSpline_3659312136439
// MI455X (gfx1250) — matrix operand order repaired and checked by reading, not compiled, not run on hardware
//
#include <hip/hip_runtime.h>
#include <hip/hip_fp16.h>

typedef __attribute__((ext_vector_type(16))) _Float16 v16h;
typedef __attribute__((ext_vector_type(8)))  float    v8f;

#define BN_EPS 1e-5f
#define NBATCH 64

__device__ __forceinline__ v8f wmma_f16(v16h a, v16h b, v8f c) {
  // v_wmma_f32_16x16x32_f16: (neg_a, A, neg_b, B, c_mod, C, reuse_a, reuse_b)
  return __builtin_amdgcn_wmma_f32_16x16x32_f16(false, a, false, b, (short)0, c,
                                                false, false);
}

// ---------------------------------------------------------------------------
// Cast conv weights f32 -> f16 (864 for conv1, 36864 for conv2..5)
// ---------------------------------------------------------------------------
__global__ void cast_weights(const float* __restrict__ c1w,
                             const float* __restrict__ cw,
                             _Float16* __restrict__ c1w_h,
                             _Float16* __restrict__ cw_h) {
  int i = blockIdx.x * 256 + threadIdx.x;
  if (i < 864)   c1w_h[i] = (_Float16)c1w[i];
  if (i < 36864) cw_h[i]  = (_Float16)cw[i];
}

// ---------------------------------------------------------------------------
// Conv1: 3ch f32 input -> 32ch f16, stride 2, K = 27 padded to one K=32 WMMA
// One wave computes a 32(out-ch) x 16(pixel) tile.
// ---------------------------------------------------------------------------
__global__ void conv1_wmma(const float* __restrict__ x,
                           const _Float16* __restrict__ w,   // [32][27]
                           const float* __restrict__ bias,   // [32]
                           const float* __restrict__ g, const float* __restrict__ bb,
                           const float* __restrict__ mm, const float* __restrict__ vv,
                           _Float16* __restrict__ y, int ntiles) {
  const int Hi = 255, Wi = 255, Ho = 127, Wo = 127;
  int wave = threadIdx.x >> 5;
  int lane = threadIdx.x & 31;
  int tile = blockIdx.x * (blockDim.x >> 5) + wave;
  if (tile >= ntiles) return;               // wave-uniform: EXEC stays all-1s
  int lo = lane & 15, hi = lane >> 4;
  int p  = tile * 16 + lo;                  // flattened (b, oh, ow)
  int hw = Ho * Wo;
  int bi = p / hw, rem = p - bi * hw;
  int oh = rem / Wo, ow = rem - oh * Wo;

  v16h a0, a1, bf;
#pragma unroll
  for (int e = 0; e < 16; ++e) {
    // A 16x32 f16 layout: lane<16 holds K {0..7,16..23}; lane>=16 {8..15,24..31}
    int ka = e + (e < 8 ? 0 : 8) + (hi ? 8 : 0);
    // B 32x16 layout: lane<16 holds K {0..7,16..23}; lane>=16 {8..15,24..31}
    int kb = e + (e < 8 ? 0 : 8) + (hi ? 8 : 0);
    a0[e] = (ka < 27) ? w[lo * 27 + ka]        : (_Float16)0;
    a1[e] = (ka < 27) ? w[(lo + 16) * 27 + ka] : (_Float16)0;
    _Float16 bv = (_Float16)0;
    if (kb < 27) {
      int ci = kb / 9, r = kb - ci * 9;
      int kh = r / 3,  kw = r - kh * 3;
      bv = (_Float16)x[((bi * 3 + ci) * Hi + (oh * 2 + kh)) * Wi + (ow * 2 + kw)];
    }
    bf[e] = bv;
  }
  v8f c0 = {}, c1 = {};
  c0 = wmma_f16(a0, bf, c0);
  c1 = wmma_f16(a1, bf, c1);

#pragma unroll
  for (int v = 0; v < 8; ++v) {
    int mrow = v + (hi ? 8 : 0);
#pragma unroll
    for (int m = 0; m < 2; ++m) {
      int co = m * 16 + mrow;
      float val = (m == 0) ? c0[v] : c1[v];
      val = fmaxf(val + bias[co], 0.0f);                      // relu(conv+b)
      float inv = g[co] * rsqrtf(vv[co] + BN_EPS);            // BN after relu
      val = (val - mm[co]) * inv + bb[co];
      y[((bi * 32 + co) * Ho + oh) * Wo + ow] = (_Float16)val;
    }
  }
}

// ---------------------------------------------------------------------------
// Conv layers 2..5: 32ch f16 -> 32ch f16, stride 2, K = 32*9 -> 9 WMMA steps.
// Weights (32*288 f16 = 18KB) staged in LDS per block.
// ---------------------------------------------------------------------------
__global__ void convn_wmma(const _Float16* __restrict__ x,
                           const _Float16* __restrict__ w,   // [32][32][3][3]
                           const float* __restrict__ bias,
                           const float* __restrict__ g, const float* __restrict__ bb,
                           const float* __restrict__ mm, const float* __restrict__ vv,
                           _Float16* __restrict__ y,
                           int Hi, int Wi, int Ho, int Wo, int ntiles) {
  __shared__ _Float16 wl[32 * 288];
  {
    const uint32_t* src = (const uint32_t*)w;
    uint32_t* dst = (uint32_t*)wl;
    for (int i = threadIdx.x; i < (32 * 288) / 2; i += blockDim.x) dst[i] = src[i];
  }
  __syncthreads();

  int wave = threadIdx.x >> 5;
  int lane = threadIdx.x & 31;
  int tile = blockIdx.x * (blockDim.x >> 5) + wave;
  if (tile >= ntiles) return;
  int lo = lane & 15, hi = lane >> 4;
  int p  = tile * 16 + lo;
  int hw = Ho * Wo;
  int bi = p / hw, rem = p - bi * hw;
  int oh = rem / Wo, ow = rem - oh * Wo;

  v8f c0 = {}, c1 = {};
#pragma unroll
  for (int s = 0; s < 9; ++s) {             // 3x3 taps, K=32 channels each
    int kh = s / 3, kw = s - kh * 3;
    v16h a0, a1, bf;
#pragma unroll
    for (int e = 0; e < 16; ++e) {
      int cia = e + (e < 8 ? 0 : 8) + (hi ? 8 : 0);   // A-fragment K (in-ch)
      int cib = e + (e < 8 ? 0 : 8) + (hi ? 8 : 0);   // B-fragment K (in-ch)
      a0[e] = wl[lo * 288 + cia * 9 + s];
      a1[e] = wl[(lo + 16) * 288 + cia * 9 + s];
      bf[e] = x[((bi * 32 + cib) * Hi + (oh * 2 + kh)) * Wi + (ow * 2 + kw)];
    }
    c0 = wmma_f16(a0, bf, c0);
    c1 = wmma_f16(a1, bf, c1);
  }

#pragma unroll
  for (int v = 0; v < 8; ++v) {
    int mrow = v + (hi ? 8 : 0);
#pragma unroll
    for (int m = 0; m < 2; ++m) {
      int co = m * 16 + mrow;
      float val = (m == 0) ? c0[v] : c1[v];
      val = fmaxf(val + bias[co], 0.0f);
      float inv = g[co] * rsqrtf(vv[co] + BN_EPS);
      val = (val - mm[co]) * inv + bb[co];
      y[((bi * 32 + co) * Ho + oh) * Wo + ow] = (_Float16)val;
    }
  }
}

// ---------------------------------------------------------------------------
// FC1(1568->20,relu) + FC2(20->10) + natural-spline Thomas solve + a,b,c coefs
// One block (1 wave) per batch row. Tiny: latency-irrelevant.
// coef layout: a = coef[0..575], b = coef[576..1151], c = coef[1152..1727]
// ---------------------------------------------------------------------------
__global__ void fc_spline(const _Float16* __restrict__ act5,  // [64][1568]
                          const float* __restrict__ l1w, const float* __restrict__ l1b,
                          const float* __restrict__ l2w, const float* __restrict__ l2b,
                          float* __restrict__ coef) {
  int b = blockIdx.x;
  int t = threadIdx.x;
  __shared__ float z1[20];
  __shared__ float ys[10];
  const _Float16* row = act5 + (size_t)b * 1568;
  if (t < 20) {
    float s = 0.f;
    for (int j = 0; j < 1568; ++j) s += (float)row[j] * l1w[t * 1568 + j];
    z1[t] = fmaxf(s + l1b[t], 0.f);
  }
  __syncthreads();
  if (t < 10) {
    float s = 0.f;
#pragma unroll
    for (int j = 0; j < 20; ++j) s += l2w[t * 20 + j] * z1[j];
    ys[t] = s + l2b[t];
  }
  __syncthreads();
  if (t == 0) {
    // Natural cubic spline: solve tridiag(1,4,1) z = (6/h^2) * d2(ys), h=1/9
    float M[10]; M[0] = 0.f; M[9] = 0.f;
    float d[8], cp[8];
    for (int i = 0; i < 8; ++i)
      d[i] = 486.0f * (ys[i] - 2.f * ys[i + 1] + ys[i + 2]);   // 6/h^2 = 486
    cp[0] = 0.25f; d[0] *= 0.25f;
    for (int i = 1; i < 8; ++i) {
      float inv = 1.0f / (4.0f - cp[i - 1]);
      cp[i] = inv;
      d[i] = (d[i] - d[i - 1]) * inv;
    }
    M[8] = d[7];
    for (int i = 6; i >= 0; --i) { d[i] = d[i] - cp[i] * d[i + 1]; M[i + 1] = d[i]; }
    float* ca = coef;
    float* cb = coef + 576;
    float* cc = coef + 1152;
    for (int i = 0; i < 9; ++i) {
      ca[b * 9 + i] = (M[i + 1] - M[i]) * 1.5f;                 // 1/(6h) = 1.5
      cb[b * 9 + i] = M[i] * 0.5f;
      cc[b * 9 + i] = 9.0f * (ys[i + 1] - ys[i])
                    - (M[i + 1] + 2.f * M[i]) * (1.0f / 54.0f); // h/6 = 1/54
    }
  }
}

// ---------------------------------------------------------------------------
// Streaming per-pixel cubic evaluation (bandwidth-bound: 50MB in + 50MB out)
// ---------------------------------------------------------------------------
__global__ void spline_eval(const float* __restrict__ x,
                            const float* __restrict__ coef,
                            float* __restrict__ out, int total) {
  int i = blockIdx.x * blockDim.x + threadIdx.x;
  if (i >= total) return;
  __builtin_prefetch(x + i + 16384, 0, 0);    // global_prefetch_b8
  float v = x[i];
  int b = i / 195075;                          // 3*255*255 per batch row
  float t = fminf(fmaxf(v * 9.0f, 0.0f), 8.0f);
  int xi = (int)floorf(t);
  float xf = v - (float)xi * (1.0f / 9.0f);
  int ix = b * 9 + xi;
  float A = coef[ix], B = coef[576 + ix], C = coef[1152 + ix];
  out[i] = ((A * xf + B) * xf + C) * xf;
}

// ---------------------------------------------------------------------------
extern "C" void kernel_launch(void* const* d_in, const int* in_sizes, int n_in,
                              void* d_out, int out_size, void* d_ws, size_t ws_size,
                              hipStream_t stream) {
  const float* batch = (const float*)d_in[0];
  const float* c1_w  = (const float*)d_in[1];
  const float* c1_b  = (const float*)d_in[2];
  const float* cw    = (const float*)d_in[3];
  const float* cb    = (const float*)d_in[4];
  const float* bn_g  = (const float*)d_in[5];
  const float* bn_b  = (const float*)d_in[6];
  const float* bn_m  = (const float*)d_in[7];
  const float* bn_v  = (const float*)d_in[8];
  const float* l1_w  = (const float*)d_in[9];
  const float* l1_b  = (const float*)d_in[10];
  const float* l2_w  = (const float*)d_in[11];
  const float* l2_b  = (const float*)d_in[12];
  float* out = (float*)d_out;

  char* ws = (char*)d_ws;
  size_t off = 0;
  auto take = [&](size_t bytes) -> char* {
    char* p = ws + off;
    off = (off + bytes + 255) & ~(size_t)255;
    return p;
  };
  _Float16* c1w_h = (_Float16*)take(864 * 2);
  _Float16* cw_h  = (_Float16*)take(36864 * 2);
  _Float16* act1  = (_Float16*)take((size_t)NBATCH * 32 * 127 * 127 * 2);
  _Float16* act2  = (_Float16*)take((size_t)NBATCH * 32 * 63 * 63 * 2);
  _Float16* act3  = (_Float16*)take((size_t)NBATCH * 32 * 31 * 31 * 2);
  _Float16* act4  = (_Float16*)take((size_t)NBATCH * 32 * 15 * 15 * 2);
  _Float16* act5  = (_Float16*)take((size_t)NBATCH * 32 * 7 * 7 * 2);
  float*    coef  = (float*)take((size_t)3 * NBATCH * 9 * 4);

  cast_weights<<<144, 256, 0, stream>>>(c1_w, cw, c1w_h, cw_h);

  {
    int ntiles = (NBATCH * 127 * 127) / 16;              // 64516, exact
    conv1_wmma<<<(ntiles + 7) / 8, 256, 0, stream>>>(
        batch, c1w_h, c1_b, bn_g, bn_b, bn_m, bn_v, act1, ntiles);
  }

  const _Float16* lin[4]  = {act1, act2, act3, act4};
  _Float16*       lout[4] = {act2, act3, act4, act5};
  const int Hs[5] = {127, 63, 31, 15, 7};
  for (int i = 0; i < 4; ++i) {
    int Hi = Hs[i], Ho = Hs[i + 1];
    int ntiles = (NBATCH * Ho * Ho) / 16;                // all exact multiples
    convn_wmma<<<(ntiles + 7) / 8, 256, 0, stream>>>(
        lin[i], cw_h + (size_t)i * 9216, cb + i * 32,
        bn_g + (i + 1) * 32, bn_b + (i + 1) * 32,
        bn_m + (i + 1) * 32, bn_v + (i + 1) * 32,
        lout[i], Hi, Hi, Ho, Ho, ntiles);
  }

  fc_spline<<<NBATCH, 32, 0, stream>>>(act5, l1_w, l1_b, l2_w, l2_b, coef);

  int total = out_size;                                   // 64*3*255*255
  spline_eval<<<(total + 255) / 256, 256, 0, stream>>>(batch, coef, out, total);
}
